// SingleMultiHeadAttention_54511724921575
// MI455X (gfx1250) — matrix operand order repaired and checked by reading, not compiled, not run on hardware
//
#include <hip/hip_runtime.h>

typedef __attribute__((ext_vector_type(16))) __bf16 v16bf;
typedef __attribute__((ext_vector_type(8)))  __bf16 v8bf;
typedef __attribute__((ext_vector_type(8)))  float  v8f;
typedef __attribute__((ext_vector_type(4)))  float  v4f;

#define NH     6
#define HD     64
#define CEMB   384
#define TSEQ   256
#define BBATCH 256
#define MROWS  (BBATCH * TSEQ)   // 65536

__device__ __forceinline__ v8f vzero8() {
  v8f z = {0.f, 0.f, 0.f, 0.f, 0.f, 0.f, 0.f, 0.f};
  return z;
}

__device__ __forceinline__ v8f wmma_bf(v16bf a, v16bf b, v8f c) {
  // D = A(16x32 bf16) * B(32x16 bf16) + C(16x16 f32)
  return __builtin_amdgcn_wmma_f32_16x16x32_bf16(false, a, false, b, (short)0, c,
                                                 false, false);
}

// A-fragment (16x32, 16-bit) from a row-major bf16 matrix.
// Contiguous-k order (B uses the same): lane holds row M=lane%16; elements 0..15 = K in [klo,klo+16)
// with klo = k0 + (lane>=16 ? 16 : 0).
// Caller passes row pointer and klo (already including the half offset).
__device__ __forceinline__ v16bf a_frag_bf(const __bf16* row, int klo) {
  v8bf lo = *(const v8bf*)(row + klo);
  v8bf hi = *(const v8bf*)(row + klo + 8);
  return __builtin_shufflevector(lo, hi, 0, 1, 2, 3, 4, 5, 6, 7,
                                 8, 9, 10, 11, 12, 13, 14, 15);
}

// Same A-fragment but sourced from fp32 with on-the-fly bf16 conversion.
__device__ __forceinline__ v16bf a_frag_f32(const float* row, int klo) {
  v4f c0 = *(const v4f*)(row + klo);
  v4f c1 = *(const v4f*)(row + klo + 4);
  v4f c2 = *(const v4f*)(row + klo + 8);
  v4f c3 = *(const v4f*)(row + klo + 12);
  v16bf a;
#pragma unroll
  for (int i = 0; i < 4; i++) {
    a[i]      = (__bf16)c0[i];
    a[4 + i]  = (__bf16)c1[i];
    a[8 + i]  = (__bf16)c2[i];
    a[12 + i] = (__bf16)c3[i];
  }
  return a;
}

// W[K,N] fp32 -> WT[N,K] bf16 so B-fragments are contiguous-K per lane.
__global__ void k_transpose_bf16(const float* __restrict__ W,
                                 __bf16* __restrict__ WT, int K, int N) {
  int i = blockIdx.x * blockDim.x + threadIdx.x;
  if (i < K * N) {
    int k = i / N, n = i - k * N;
    WT[(size_t)n * K + k] = (__bf16)W[i];
  }
}

// qkv = x @ W_qkv, scattered into Q[B,H,T,D] (pre-scaled 1/sqrt(D)),
// K[B,H,T,D], V transposed [B,H,D,T], all bf16.
__global__ void __launch_bounds__(256) k_gemm_qkv(
    const float* __restrict__ X,    // [65536, 384] fp32
    const __bf16* __restrict__ WT,  // [1152, 384] bf16 (W_qkv^T)
    __bf16* __restrict__ Q, __bf16* __restrict__ Kc, __bf16* __restrict__ Vt) {
  const int lane = threadIdx.x & 31;
  const int l16 = lane & 15;
  const int half = (lane >> 4) & 1;
  const int wv = threadIdx.x >> 5;
  const int m0 = blockIdx.x * 128 + (wv & 3) * 32;
  const int n0 = blockIdx.y * 128 + (wv >> 2) * 64;

  v8f acc[2][4];
#pragma unroll
  for (int i = 0; i < 2; i++)
#pragma unroll
    for (int j = 0; j < 4; j++) acc[i][j] = vzero8();

#pragma unroll 1
  for (int k0 = 0; k0 < CEMB; k0 += 32) {
    v16bf a[2];
#pragma unroll
    for (int mt = 0; mt < 2; mt++)
      a[mt] = a_frag_f32(X + (size_t)(m0 + mt * 16 + l16) * CEMB, k0 + half * 16);
#pragma unroll
    for (int nt = 0; nt < 4; nt++) {
      v16bf b = *(const v16bf*)(WT + (size_t)(n0 + nt * 16 + l16) * CEMB +
                                k0 + half * 16);
#pragma unroll
      for (int mt = 0; mt < 2; mt++) acc[mt][nt] = wmma_bf(a[mt], b, acc[mt][nt]);
    }
  }

  const float qscale = 0.125f;  // 1/sqrt(HD)
#pragma unroll
  for (int mt = 0; mt < 2; mt++) {
#pragma unroll
    for (int nt = 0; nt < 4; nt++) {
#pragma unroll
      for (int r = 0; r < 8; r++) {
        int m = m0 + mt * 16 + r + half * 8;   // C/D layout: row = r + 8*half
        int n = n0 + nt * 16 + l16;            // col = lane & 15
        int bq = m >> 8, t = m & 255;
        float v = acc[mt][nt][r];
        if (n < CEMB) {
          int h = n >> 6, d = n & 63;
          Q[(size_t)((bq * NH + h) * TSEQ + t) * HD + d] = (__bf16)(v * qscale);
        } else if (n < 2 * CEMB) {
          int n2 = n - CEMB, h = n2 >> 6, d = n2 & 63;
          Kc[(size_t)((bq * NH + h) * TSEQ + t) * HD + d] = (__bf16)v;
        } else {
          int n2 = n - 2 * CEMB, h = n2 >> 6, d = n2 & 63;
          Vt[(size_t)((bq * NH + h) * HD + d) * TSEQ + t] = (__bf16)v;
        }
      }
    }
  }
}

// Flash-style causal attention: one workgroup per (b,h); wave w owns query
// rows [w*32, w*32+32) and streams key blocks kb = 0..w (causal skip).
__global__ void __launch_bounds__(256) k_attn(
    const __bf16* __restrict__ Q, const __bf16* __restrict__ Kc,
    const __bf16* __restrict__ Vt, __bf16* __restrict__ O) {
  __shared__ __bf16 Plds[8][32][32];  // per-wave P staging (C-layout -> A-layout)
  const int lane = threadIdx.x & 31;
  const int l16 = lane & 15;
  const int half = (lane >> 4) & 1;
  const int wv = threadIdx.x >> 5;
  const int bh = blockIdx.x;
  const int b = bh / NH, h = bh - b * NH;
  const int row0 = wv * 32;

  const __bf16* Qh = Q + (size_t)bh * TSEQ * HD;
  const __bf16* Kh = Kc + (size_t)bh * TSEQ * HD;
  const __bf16* Vh = Vt + (size_t)bh * HD * TSEQ;

  // Q fragments held in registers for the whole pass (rows x D=64 -> 2x2 frags)
  v16bf aq[2][2];
#pragma unroll
  for (int mt = 0; mt < 2; mt++)
#pragma unroll
    for (int ks = 0; ks < 2; ks++)
      aq[mt][ks] =
          a_frag_bf(Qh + (size_t)(row0 + mt * 16 + l16) * HD, ks * 32 + half * 16);

  v8f acc[2][4];
  float ms[2][8], ls[2][8];
#pragma unroll
  for (int mt = 0; mt < 2; mt++) {
#pragma unroll
    for (int j = 0; j < 4; j++) acc[mt][j] = vzero8();
#pragma unroll
    for (int r = 0; r < 8; r++) { ms[mt][r] = -3.0e38f; ls[mt][r] = 0.f; }
  }

  for (int kb = 0; kb <= wv; kb++) {
    const int s0 = kb * 32;
    // S = Q * K^T for this 32-key block
    v8f s[2][2];
#pragma unroll
    for (int nt = 0; nt < 2; nt++) {
      v16bf bk0 = *(const v16bf*)(Kh + (size_t)(s0 + nt * 16 + l16) * HD + half * 16);
      v16bf bk1 = *(const v16bf*)(Kh + (size_t)(s0 + nt * 16 + l16) * HD + 32 + half * 16);
#pragma unroll
      for (int mt = 0; mt < 2; mt++) {
        v8f tmp = wmma_bf(aq[mt][0], bk0, vzero8());
        s[mt][nt] = wmma_bf(aq[mt][1], bk1, tmp);
      }
    }
    // causal mask + online softmax (row = r + 8*half across 16 lanes)
#pragma unroll
    for (int mt = 0; mt < 2; mt++) {
#pragma unroll
      for (int r = 0; r < 8; r++) {
        int tq = row0 + mt * 16 + r + half * 8;
        float v0 = s[mt][0][r];
        float v1 = s[mt][1][r];
        if (s0 + l16 > tq) v0 = -1.0e30f;
        if (s0 + 16 + l16 > tq) v1 = -1.0e30f;
        float mx = fmaxf(v0, v1);
        mx = fmaxf(mx, __shfl_xor(mx, 1));
        mx = fmaxf(mx, __shfl_xor(mx, 2));
        mx = fmaxf(mx, __shfl_xor(mx, 4));
        mx = fmaxf(mx, __shfl_xor(mx, 8));
        float mnew = fmaxf(ms[mt][r], mx);
        float alpha = __expf(ms[mt][r] - mnew);
        ms[mt][r] = mnew;
        float p0 = __expf(v0 - mnew);
        float p1 = __expf(v1 - mnew);
        float rs = p0 + p1;
        rs += __shfl_xor(rs, 1);
        rs += __shfl_xor(rs, 2);
        rs += __shfl_xor(rs, 4);
        rs += __shfl_xor(rs, 8);
        ls[mt][r] = ls[mt][r] * alpha + rs;
#pragma unroll
        for (int nd = 0; nd < 4; nd++) acc[mt][nd][r] *= alpha;
        Plds[wv][mt * 16 + r + half * 8][l16] = (__bf16)p0;
        Plds[wv][mt * 16 + r + half * 8][16 + l16] = (__bf16)p1;
      }
    }
    // O += P * V  (P re-read from LDS in A-fragment layout; per-wave LDS is
    // in-order, so same-wave RAW through LDS needs no barrier)
    v16bf pa[2];
#pragma unroll
    for (int mt = 0; mt < 2; mt++)
      pa[mt] = a_frag_bf(&Plds[wv][mt * 16 + l16][0], half * 16);
#pragma unroll
    for (int nd = 0; nd < 4; nd++) {
      v16bf bv = *(const v16bf*)(Vh + (size_t)(nd * 16 + l16) * TSEQ + s0 + half * 16);
#pragma unroll
      for (int mt = 0; mt < 2; mt++) acc[mt][nd] = wmma_bf(pa[mt], bv, acc[mt][nd]);
    }
  }

  // normalize and scatter to [B,T,C] bf16 (c = h*64 + d)
#pragma unroll
  for (int mt = 0; mt < 2; mt++) {
#pragma unroll
    for (int r = 0; r < 8; r++) {
      float inv = 1.0f / ls[mt][r];
      int tq = row0 + mt * 16 + r + half * 8;
#pragma unroll
      for (int nd = 0; nd < 4; nd++) {
        int d = nd * 16 + l16;
        O[(size_t)(b * TSEQ + tq) * CEMB + h * HD + d] = (__bf16)(acc[mt][nd][r] * inv);
      }
    }
  }
}

// y = AttOut(bf16) @ W_proj + b_proj, fp32 output.
__global__ void __launch_bounds__(256) k_gemm_proj(
    const __bf16* __restrict__ A,   // [65536, 384] bf16
    const __bf16* __restrict__ WT,  // [384, 384] bf16 (W_proj^T)
    const float* __restrict__ bias, float* __restrict__ Y) {
  const int lane = threadIdx.x & 31;
  const int l16 = lane & 15;
  const int half = (lane >> 4) & 1;
  const int wv = threadIdx.x >> 5;
  const int m0 = blockIdx.x * 128 + (wv & 3) * 32;
  const int n0 = blockIdx.y * 128 + (wv >> 2) * 64;

  v8f acc[2][4];
#pragma unroll
  for (int i = 0; i < 2; i++)
#pragma unroll
    for (int j = 0; j < 4; j++) acc[i][j] = vzero8();

#pragma unroll 1
  for (int k0 = 0; k0 < CEMB; k0 += 32) {
    v16bf a[2];
#pragma unroll
    for (int mt = 0; mt < 2; mt++)
      a[mt] = a_frag_bf(A + (size_t)(m0 + mt * 16 + l16) * CEMB, k0 + half * 16);
#pragma unroll
    for (int nt = 0; nt < 4; nt++) {
      v16bf bm = *(const v16bf*)(WT + (size_t)(n0 + nt * 16 + l16) * CEMB +
                                 k0 + half * 16);
#pragma unroll
      for (int mt = 0; mt < 2; mt++) acc[mt][nt] = wmma_bf(a[mt], bm, acc[mt][nt]);
    }
  }

#pragma unroll
  for (int nt = 0; nt < 4; nt++) {
    float bb = bias[n0 + nt * 16 + l16];
#pragma unroll
    for (int mt = 0; mt < 2; mt++) {
#pragma unroll
      for (int r = 0; r < 8; r++) {
        int m = m0 + mt * 16 + r + half * 8;
        Y[(size_t)m * CEMB + n0 + nt * 16 + l16] = acc[mt][nt][r] + bb;
      }
    }
  }
}

extern "C" void kernel_launch(void* const* d_in, const int* in_sizes, int n_in,
                              void* d_out, int out_size, void* d_ws, size_t ws_size,
                              hipStream_t stream) {
  (void)in_sizes; (void)n_in; (void)out_size; (void)ws_size;
  const float* x      = (const float*)d_in[0];  // [256,256,384]
  const float* W_qkv  = (const float*)d_in[1];  // [384,1152]
  const float* W_proj = (const float*)d_in[2];  // [384,384]
  const float* b_proj = (const float*)d_in[3];  // [384]
  float* out = (float*)d_out;

  char* ws = (char*)d_ws;
  size_t off = 0;
  auto carve = [&](size_t bytes) -> char* {
    char* p = ws + off;
    off += (bytes + 255) & ~(size_t)255;
    return p;
  };
  __bf16* Q   = (__bf16*)carve((size_t)BBATCH * NH * TSEQ * HD * 2);
  __bf16* Kc  = (__bf16*)carve((size_t)BBATCH * NH * TSEQ * HD * 2);
  __bf16* Vt  = (__bf16*)carve((size_t)BBATCH * NH * HD * TSEQ * 2);
  __bf16* AO  = (__bf16*)carve((size_t)MROWS * CEMB * 2);
  __bf16* WqT = (__bf16*)carve((size_t)3 * CEMB * CEMB * 2);
  __bf16* WpT = (__bf16*)carve((size_t)CEMB * CEMB * 2);

  {
    int tot = CEMB * 3 * CEMB;
    k_transpose_bf16<<<(tot + 255) / 256, 256, 0, stream>>>(W_qkv, WqT, CEMB, 3 * CEMB);
  }
  {
    int tot = CEMB * CEMB;
    k_transpose_bf16<<<(tot + 255) / 256, 256, 0, stream>>>(W_proj, WpT, CEMB, CEMB);
  }
  k_gemm_qkv<<<dim3(MROWS / 128, (3 * CEMB) / 128), 256, 0, stream>>>(x, WqT, Q, Kc, Vt);
  k_attn<<<dim3(BBATCH * NH), 256, 0, stream>>>(Q, Kc, Vt, AO);
  k_gemm_proj<<<dim3(MROWS / 128, CEMB / 128), 256, 0, stream>>>(AO, WpT, b_proj, out);
}
